// FusedMoEMethod_5437428596961
// MI455X (gfx1250) — compile-verified, matrix operand order checked by reading, not run on hardware
//
#include <hip/hip_runtime.h>
#include <stdint.h>

// Problem constants (from reference)
constexpr int TT = 1024;   // tokens
constexpr int HH = 2048;   // hidden
constexpr int II = 1408;   // intermediate
constexpr int EE = 8;      // experts
constexpr int KK = 2;      // top-k
constexpr int ROWCAP = TT * KK + EE * 32;  // 2304 compacted rows incl. 32-row tile padding

typedef __attribute__((ext_vector_type(16))) __bf16    v16bf;
typedef __attribute__((ext_vector_type(16))) float     v16f;
typedef __attribute__((ext_vector_type(8)))  float     v8f;
typedef __attribute__((ext_vector_type(8)))  uint32_t  v8u;

union BU { v16bf b; v8u u; };

// ---- helpers -------------------------------------------------------------

__device__ __forceinline__ __bf16 f2bf(float f) {     // RNE, epilogue only
  union { float f; uint32_t u; } v; v.f = f;
  uint32_t r = (v.u + 0x7FFFu + ((v.u >> 16) & 1u)) >> 16;
  union { uint16_t s; __bf16 b; } o; o.s = (uint16_t)r;
  return o.b;
}

// Hot-loop conversion: 16 f32 -> 16 bf16 via 8x v_perm_b32 (truncate).
__device__ __forceinline__ v16bf cvtB(const float* __restrict__ p) {
  v16f f = *(const v16f*)p;           // 64B aligned by construction
  union { v8u u; v16bf b; } o;
#pragma unroll
  for (int i = 0; i < 8; ++i) {
    union { float f; uint32_t u; } a0, a1;
    a0.f = f[2 * i];                  // -> low bf16
    a1.f = f[2 * i + 1];              // -> high bf16
    o.u[i] = __builtin_amdgcn_perm(a1.u, a0.u, 0x7060302u);
  }
  return o.b;
}

__device__ __forceinline__ v8f wmma_bf16(v16bf a, v16bf b, v8f c) {
  return __builtin_amdgcn_wmma_f32_16x16x32_bf16(
      /*neg_a=*/false, a, /*neg_b=*/false, b,
      /*c_mod=*/(short)0, c, /*reuse_a=*/false, /*reuse_b=*/false);
}

// ---- kernel 1: init (zero out, convert x -> bf16, zero bookkeeping) ------

__global__ void moe_init(const float* __restrict__ x, __bf16* __restrict__ xb,
                         float* __restrict__ out, int* __restrict__ counts,
                         int* __restrict__ row_token, float* __restrict__ row_w) {
  int i = blockIdx.x * blockDim.x + threadIdx.x;
  const int N = TT * HH;
  if (i < N) {
    out[i] = 0.0f;
    xb[i]  = f2bf(x[i]);
  }
  if (i < EE) counts[i] = 0;
  if (i < ROWCAP) { row_token[i] = 0; row_w[i] = 0.0f; }
}

// ---- kernel 2: router (softmax -> top2 -> renorm) ------------------------

__global__ void moe_router(const float* __restrict__ logits,
                           int* __restrict__ counts,
                           int* __restrict__ ids, float* __restrict__ wts) {
  int t = blockIdx.x * blockDim.x + threadIdx.x;
  if (t >= TT) return;
  float l[EE];
  float mx = -1e30f;
#pragma unroll
  for (int e = 0; e < EE; ++e) { l[e] = logits[t * EE + e]; mx = fmaxf(mx, l[e]); }
  float s = 0.0f;
#pragma unroll
  for (int e = 0; e < EE; ++e) { l[e] = __expf(l[e] - mx); s += l[e]; }
  int i0 = 0;
#pragma unroll
  for (int e = 1; e < EE; ++e) if (l[e] > l[i0]) i0 = e;
  int i1 = (i0 == 0) ? 1 : 0;
#pragma unroll
  for (int e = 0; e < EE; ++e) if (e != i0 && l[e] > l[i1]) i1 = e;
  float p0 = l[i0] / s, p1 = l[i1] / s;
  float r = 1.0f / (p0 + p1);          // renormalize over top-k
  ids[t * 2 + 0] = i0; ids[t * 2 + 1] = i1;
  wts[t * 2 + 0] = p0 * r; wts[t * 2 + 1] = p1 * r;
  atomicAdd(&counts[i0], 1);
  atomicAdd(&counts[i1], 1);
}

// ---- kernel 3: tiny scan (padded-to-32 offsets) --------------------------

__global__ void moe_scan(const int* __restrict__ counts,
                         int* __restrict__ aoff, int* __restrict__ cursors) {
  if (threadIdx.x == 0 && blockIdx.x == 0) {
    int acc = 0;
    for (int e = 0; e < EE; ++e) {
      aoff[e] = acc;
      acc += (counts[e] + 31) & ~31;
      cursors[e] = 0;
    }
    aoff[EE] = acc;
  }
}

// ---- kernel 4: scatter tokens into per-expert compacted lists ------------

__global__ void moe_scatter(const int* __restrict__ ids, const float* __restrict__ wts,
                            const int* __restrict__ aoff, int* __restrict__ cursors,
                            int* __restrict__ row_token, float* __restrict__ row_w) {
  int t = blockIdx.x * blockDim.x + threadIdx.x;
  if (t >= TT) return;
#pragma unroll
  for (int k = 0; k < KK; ++k) {
    int e = ids[t * 2 + k];
    int pos = atomicAdd(&cursors[e], 1);
    int slot = aoff[e] + pos;
    row_token[slot] = t;
    row_w[slot] = wts[t * 2 + k];
  }
}

// ---- kernel 5: grouped GEMM1 + SiLU*up ------------------------------------
// Block = 4 waves, block tile 128 rows x 64 cols of I.  The 64 B rows
// (gate frags 0-3, up frags 4-7) are converted f32->bf16 cooperatively and
// staged in double-buffered LDS once per k-step.  Each wave preloads B
// fragments in groups of 4 (live array -> clustered ds_loads, one wait,
// 8 back-to-back WMMAs) on its own 32-row m-subtile.
// grid = (II/64, TT/128, EE), block = 128.

__global__ __launch_bounds__(128) void moe_gemm1(
    const float* __restrict__ w13,       // [E, 2I, H] row-major
    const __bf16* __restrict__ xb,       // [T, H]
    const int* __restrict__ aoff,
    const int* __restrict__ row_token,
    __bf16* __restrict__ act) {          // [ROWCAP, I]
  __shared__ __align__(32) v8u sB[2 * 8 * 32];   // 16 KB, double buffered

  int e = blockIdx.z;
  int base = aoff[e];
  int rows = aoff[e + 1] - base;         // padded row count for this expert
  if ((int)blockIdx.y * 128 >= rows) return;     // whole block out of range

  int tid  = threadIdx.x;
  int wid  = tid >> 5;                   // wave 0..3
  int lane = tid & 31;
  int hk   = lane >> 4;                  // 0: K 0..15, 1: K 16..31
  int l15  = lane & 15;
  int n0   = blockIdx.x * 64;
  int m0   = blockIdx.y * 128 + wid * 32;        // this wave's m-subtile
  bool active = (m0 < rows);             // wave-uniform

  const float* wg = w13 + (size_t)e * (2 * II) * HH;

  // staging pointers: this wave stages fragments 2*wid and 2*wid+1
  const float* sp[2];
#pragma unroll
  for (int j = 0; j < 2; ++j) {
    int f = 2 * wid + j;
    int rowi = (f < 4) ? (n0 + f * 16) : (II + n0 + (f - 4) * 16);
    sp[j] = wg + (size_t)(rowi + l15) * HH + hk * 16;
  }

  const __bf16* arow0 = xb;
  const __bf16* arow1 = xb;
  if (active) {
    int tok0 = row_token[base + m0 + l15];
    int tok1 = row_token[base + m0 + 16 + l15];
    arow0 = xb + (size_t)tok0 * HH + hk * 16;
    arow1 = xb + (size_t)tok1 * HH + hk * 16;
  }

  v8f acc0[8] = {}, acc1[8] = {};        // [0..3] gate, [4..7] up

  // prologue: stage k=0 into buffer 0
#pragma unroll
  for (int j = 0; j < 2; ++j) {
    BU t; t.b = cvtB(sp[j]);
    sB[(2 * wid + j) * 32 + lane] = t.u;
  }
  __syncthreads();

  const int nk = HH / 32;
#pragma unroll 1
  for (int kk = 0; kk < nk; ++kk) {
    int k = kk * 32;
    int cur = (kk & 1) * 256;
    // stage next k-chunk into the other buffer (overlaps with WMMAs below)
    if (kk + 1 < nk) {
      int nxt = ((kk + 1) & 1) * 256;
#pragma unroll
      for (int j = 0; j < 2; ++j) {
        __builtin_prefetch((const void*)(sp[j] + k + 96), 0, 0);
        BU t; t.b = cvtB(sp[j] + k + 32);
        sB[nxt + (2 * wid + j) * 32 + lane] = t.u;
      }
    }
    if (active) {
      v16bf a0 = *(const v16bf*)(arow0 + k);
      v16bf a1 = *(const v16bf*)(arow1 + k);
      // group 0: fragments 0-3 preloaded together -> clustered ds_loads
      BU t[4];
#pragma unroll
      for (int j = 0; j < 4; ++j) t[j].u = sB[cur + j * 32 + lane];
#pragma unroll
      for (int j = 0; j < 4; ++j) {
        acc0[j] = wmma_bf16(a0, t[j].b, acc0[j]);
        acc1[j] = wmma_bf16(a1, t[j].b, acc1[j]);
      }
      // group 1: fragments 4-7; loads issue under group-0 WMMA shadow
#pragma unroll
      for (int j = 0; j < 4; ++j) t[j].u = sB[cur + (4 + j) * 32 + lane];
#pragma unroll
      for (int j = 0; j < 4; ++j) {
        acc0[4 + j] = wmma_bf16(a0, t[j].b, acc0[4 + j]);
        acc1[4 + j] = wmma_bf16(a1, t[j].b, acc1[4 + j]);
      }
    }
    __syncthreads();
  }

  if (!active) return;
  // epilogue: act = silu(gate) * up   (gate/up accumulator layouts coincide)
  int srow = base + m0;
#pragma unroll
  for (int f = 0; f < 4; ++f) {
#pragma unroll
    for (int r = 0; r < 8; ++r) {
      int col = n0 + f * 16 + l15;
      {
        float g = acc0[f][r], u = acc0[f + 4][r];
        float sg = g / (1.0f + __expf(-g));
        act[(size_t)(srow + r + hk * 8) * II + col] = f2bf(sg * u);
      }
      {
        float g = acc1[f][r], u = acc1[f + 4][r];
        float sg = g / (1.0f + __expf(-g));
        act[(size_t)(srow + 16 + r + hk * 8) * II + col] = f2bf(sg * u);
      }
    }
  }
}

// ---- kernel 6: grouped GEMM2, weighted scatter-add into out --------------
// Same LDS-staged scheme: block tile 128 rows x 64 cols of H, 4 fragments,
// each wave stages one; all 4 fragments preloaded as a live group.
// grid = (HH/64, TT/128, EE), block = 128.

__global__ __launch_bounds__(128) void moe_gemm2(
    const float* __restrict__ w2,        // [E, H, I] row-major
    const __bf16* __restrict__ act,      // [ROWCAP, I]
    const int* __restrict__ aoff,
    const int* __restrict__ counts,
    const int* __restrict__ row_token,
    const float* __restrict__ row_w,
    float* __restrict__ out) {           // [T, H]
  __shared__ __align__(32) v8u sB[2 * 4 * 32];   // 8 KB, double buffered

  int e = blockIdx.z;
  int base = aoff[e];
  int rows = aoff[e + 1] - base;
  int cnt = counts[e];
  if ((int)blockIdx.y * 128 >= rows) return;

  int tid  = threadIdx.x;
  int wid  = tid >> 5;
  int lane = tid & 31;
  int hk   = lane >> 4;
  int l15  = lane & 15;
  int n0   = blockIdx.x * 64;
  int m0   = blockIdx.y * 128 + wid * 32;
  bool active = (m0 < rows);

  const float* w2e = w2 + (size_t)e * HH * II;
  const float* sp = w2e + (size_t)(n0 + wid * 16 + l15) * II + hk * 16;  // frag = wid

  const __bf16* arow0 = act;
  const __bf16* arow1 = act;
  if (active) {
    arow0 = act + (size_t)(base + m0 + l15) * II + hk * 16;
    arow1 = act + (size_t)(base + m0 + 16 + l15) * II + hk * 16;
  }

  v8f c0[4] = {}, c1[4] = {};

  { BU t; t.b = cvtB(sp); sB[wid * 32 + lane] = t.u; }
  __syncthreads();

  const int nk = II / 32;
#pragma unroll 1
  for (int kk = 0; kk < nk; ++kk) {
    int k = kk * 32;
    int cur = (kk & 1) * 128;
    if (kk + 1 < nk) {
      int nxt = ((kk + 1) & 1) * 128;
      __builtin_prefetch((const void*)(sp + k + 96), 0, 0);
      BU t; t.b = cvtB(sp + k + 32);
      sB[nxt + wid * 32 + lane] = t.u;
    }
    if (active) {
      v16bf a0 = *(const v16bf*)(arow0 + k);
      v16bf a1 = *(const v16bf*)(arow1 + k);
      BU t[4];
#pragma unroll
      for (int j = 0; j < 4; ++j) t[j].u = sB[cur + j * 32 + lane];
#pragma unroll
      for (int j = 0; j < 4; ++j) {
        c0[j] = wmma_bf16(a0, t[j].b, c0[j]);
        c1[j] = wmma_bf16(a1, t[j].b, c1[j]);
      }
    }
    __syncthreads();
  }

  if (!active) return;
  // epilogue: scale by routing weight, atomic-add into dense output
#pragma unroll
  for (int half = 0; half < 2; ++half) {
#pragma unroll
    for (int r = 0; r < 8; ++r) {
      int m = half * 16 + r + hk * 8;
      if (m0 + m < cnt) {
        int slot = base + m0 + m;
        int tok = row_token[slot];
        float w = row_w[slot];
#pragma unroll
        for (int f = 0; f < 4; ++f) {
          float v = (half ? c1[f][r] : c0[f][r]) * w;
          atomicAdd(&out[(size_t)tok * HH + n0 + f * 16 + l15], v);
        }
      }
    }
  }
}

// ---- workspace layout (bytes, 64B aligned slabs) -------------------------
constexpr size_t OFF_COUNTS = 0;                        // 8 int
constexpr size_t OFF_AOFF   = 64;                       // 9 int
constexpr size_t OFF_CUR    = 128;                      // 8 int
constexpr size_t OFF_IDS    = 192;                      // 2048 int
constexpr size_t OFF_TKW    = OFF_IDS + 2048 * 4;       // 2048 float
constexpr size_t OFF_RTOK   = OFF_TKW + 2048 * 4;       // ROWCAP int
constexpr size_t OFF_RW     = OFF_RTOK + ROWCAP * 4;    // ROWCAP float
constexpr size_t OFF_XB     = ((OFF_RW + ROWCAP * 4 + 127) / 128) * 128;  // T*H bf16
constexpr size_t OFF_ACT    = ((OFF_XB + (size_t)TT * HH * 2 + 127) / 128) * 128; // ROWCAP*I bf16

extern "C" void kernel_launch(void* const* d_in, const int* in_sizes, int n_in,
                              void* d_out, int out_size, void* d_ws, size_t ws_size,
                              hipStream_t stream) {
  const float* x      = (const float*)d_in[0];   // [T, H]
  const float* logits = (const float*)d_in[1];   // [T, E]
  const float* w13    = (const float*)d_in[2];   // [E, 2I, H]
  const float* w2     = (const float*)d_in[3];   // [E, H, I]
  float* out = (float*)d_out;                    // [T, H]

  char* ws = (char*)d_ws;
  int*    counts   = (int*)(ws + OFF_COUNTS);
  int*    aoff     = (int*)(ws + OFF_AOFF);
  int*    cursors  = (int*)(ws + OFF_CUR);
  int*    ids      = (int*)(ws + OFF_IDS);
  float*  tkw      = (float*)(ws + OFF_TKW);
  int*    row_tok  = (int*)(ws + OFF_RTOK);
  float*  row_w    = (float*)(ws + OFF_RW);
  __bf16* xb       = (__bf16*)(ws + OFF_XB);
  __bf16* act      = (__bf16*)(ws + OFF_ACT);

  // 1. init: zero out/bookkeeping, convert x -> bf16
  {
    int n = TT * HH;
    moe_init<<<(n + 255) / 256, 256, 0, stream>>>(x, xb, out, counts, row_tok, row_w);
  }
  // 2. router
  moe_router<<<(TT + 255) / 256, 256, 0, stream>>>(logits, counts, ids, tkw);
  // 3. scan
  moe_scan<<<1, 32, 0, stream>>>(counts, aoff, cursors);
  // 4. scatter
  moe_scatter<<<(TT + 255) / 256, 256, 0, stream>>>(ids, tkw, aoff, cursors, row_tok, row_w);
  // 5. grouped GEMM1 + SiLU  (block tile 128 rows)
  {
    dim3 g(II / 64, TT / 128, EE);
    moe_gemm1<<<g, 128, 0, stream>>>(w13, xb, aoff, row_tok, act);
  }
  // 6. grouped GEMM2 + weighted combine
  {
    dim3 g(HH / 64, TT / 128, EE);
    moe_gemm2<<<g, 128, 0, stream>>>(w2, act, aoff, counts, row_tok, row_w, out);
  }
}
